// GNNNet_85375359910684
// MI455X (gfx1250) — compile-verified, matrix operand order checked by reading, not run on hardware
//
#include <hip/hip_runtime.h>
#include <hip/hip_bf16.h>
#include <stdint.h>

#define F_IN     78
#define N_NODES  131072
#define N_EDGES  524288
#define N_GRAPHS 4096

typedef __attribute__((ext_vector_type(16))) _Float16 v16h;
typedef __attribute__((ext_vector_type(8)))  _Float16 v8h;
typedef __attribute__((ext_vector_type(4)))  _Float16 v4h;
typedef __attribute__((ext_vector_type(8)))  float    v8f;

// ---------------------------------------------------------------------------
// WMMA GEMM: C[M x ldc] = A[M x lda](fp32) * B[Kpad x Npad](f16)  (+bias)(+relu)
// M % 128 == 0, Kpad % 32 == 0, Npad % 64 == 0, lda % 4 == 0.
// Zero padding already present in A / B.
// Block: 256 threads = 8 waves; block tile 128x64; each wave owns one 16-row
// strip and all 64 columns -> 4x v_wmma_f32_16x16x32_f16 per K-step.
// ---------------------------------------------------------------------------
#define BM 128
#define BN 64
#define BK 32

__global__ void __launch_bounds__(256)
gemm_wmma_f16(const float* __restrict__ A, const _Float16* __restrict__ B,
              float* __restrict__ C, int Kpad, int Npad, int lda, int ldc,
              const float* __restrict__ bias, int do_relu)
{
    __shared__ _Float16 As[BM * BK];   // [row][k]
    __shared__ _Float16 Bs[BN * BK];   // transposed: [col][k]

    const int tid  = threadIdx.x;
    const int lane = tid & 31;
    const int wave = tid >> 5;         // 0..7 -> 16-row strip within 128

    const int m0 = blockIdx.x * BM;
    const int n0 = blockIdx.y * BN;

    v8f acc[4] = {};

    const int row_a = wave * 16 + (lane & 15);     // A frag row
    const int kb    = (lane >> 4) * 8;             // 0 or 8
    const int hi    = kb >> 3;                     // 0 or 1
    const int cbase = lane & 15;                   // B frag col base

    // A tile: 128x32 fp32 = 1024 float4; 4 per thread (row r, 16B chunk c4)
    const int a_r  = (tid >> 3);                   // +32 per i (4096/32=128 rows / 4 iters)
    const int a_c4 = (tid & 7) * 4;
    // B tile: 32x64 f16 = 256 x 8-half chunks; 1 per thread
    const int b_r  = tid >> 3;
    const int b_c8 = (tid & 7) * 8;

    for (int k0 = 0; k0 < Kpad; k0 += BK) {
        // ---- wide global loads into registers (b128, batched waits) ----
        float4 av[4];
        #pragma unroll
        for (int i = 0; i < 4; ++i) {
            int r = a_r + i * 32;
            av[i] = *(const float4*)&A[(size_t)(m0 + r) * lda + (k0 + a_c4)];
        }
        v8h bv = *(const v8h*)&B[(size_t)(k0 + b_r) * Npad + (n0 + b_c8)];

        // ---- prefetch next K panel while this one computes (global_prefetch) ----
        if (k0 + BK < Kpad) {
            __builtin_prefetch(&A[(size_t)(m0 + (tid >> 1)) * lda + (k0 + BK)], 0, 0);
            __builtin_prefetch(&B[(size_t)(k0 + BK + (tid & 31)) * Npad + n0], 0, 0);
        }

        // ---- convert + store to LDS (A: packed ds_store_b64; B: transpose) ----
        #pragma unroll
        for (int i = 0; i < 4; ++i) {
            int r = a_r + i * 32;
            v4h h;
            h[0] = (_Float16)av[i].x;
            h[1] = (_Float16)av[i].y;
            h[2] = (_Float16)av[i].z;
            h[3] = (_Float16)av[i].w;
            *(v4h*)&As[r * BK + a_c4] = h;
        }
        #pragma unroll
        for (int j = 0; j < 8; ++j)
            Bs[(b_c8 + j) * BK + b_r] = bv[j];
        __syncthreads();

        // ---- A fragment (16B LDS loads) ----
        const v8h* arow = (const v8h*)(As + row_a * BK);
        v16h a = __builtin_shufflevector(arow[hi], arow[2 + hi],
                                         0,1,2,3,4,5,6,7,8,9,10,11,12,13,14,15);

        // ---- 4 B fragments / 4 WMMAs ----
        #pragma unroll
        for (int t = 0; t < 4; ++t) {
            const v8h* brow = (const v8h*)(Bs + (cbase + 16 * t) * BK);
            v16h b = __builtin_shufflevector(brow[hi], brow[2 + hi],
                                             0,1,2,3,4,5,6,7,8,9,10,11,12,13,14,15);
            acc[t] = __builtin_amdgcn_wmma_f32_16x16x32_f16(false, a, false, b,
                                                            (short)0, acc[t], false, false);
        }
        __syncthreads();
    }

    // ---- store D: VGPR r -> M = r + 8*(lane>=16), N = lane&15 ----
    const int mbase = m0 + wave * 16 + ((lane >> 4) * 8);
    #pragma unroll
    for (int t = 0; t < 4; ++t) {
        const int ncol = n0 + 16 * t + (lane & 15);
        const float bvv = bias ? bias[ncol] : 0.0f;
        #pragma unroll
        for (int r = 0; r < 8; ++r) {
            float v = acc[t][r] + bvv;
            if (do_relu) v = fmaxf(v, 0.0f);
            C[(size_t)(mbase + r) * ldc + ncol] = v;
        }
    }
}

// ---------------------------------------------------------------------------
// Utility kernels
// ---------------------------------------------------------------------------
__global__ void fill_f32(float* p, float v, int n) {
    int i = blockIdx.x * blockDim.x + threadIdx.x;
    if (i < n) p[i] = v;
}

__global__ void deg_accum(const int* __restrict__ dst, float* __restrict__ deg, int E) {
    int e = blockIdx.x * blockDim.x + threadIdx.x;
    if (e < E) atomicAdd(&deg[dst[e]], 1.0f);
}

__global__ void make_dinv(float* deg, int n) {
    int i = blockIdx.x * blockDim.x + threadIdx.x;
    if (i < n) {
        float d = deg[i];
        deg[i] = d > 0.0f ? rsqrtf(d) : 0.0f;
    }
}

// copy x[N x F] -> X[N x ldo] zero padded
__global__ void pad_input(const float* __restrict__ x, float* __restrict__ X,
                          int F, int ldo, size_t total) {
    size_t idx = (size_t)blockIdx.x * blockDim.x + threadIdx.x;
    if (idx >= total) return;
    int f = (int)(idx % ldo);
    size_t i = idx / ldo;
    X[idx] = (f < F) ? x[i * F + f] : 0.0f;
}

// one block per edge: acc[dst] += H[src] * dinv[src]*dinv[dst]
__global__ void scatter_edges(const float* __restrict__ H, float* __restrict__ acc,
                              const int* __restrict__ src, const int* __restrict__ dst,
                              const float* __restrict__ dinv, int F4, int ld) {
    int e = blockIdx.x;
    int s = src[e], d = dst[e];
    float w = dinv[s] * dinv[d];
    const float4* hrow = (const float4*)(H + (size_t)s * ld);
    float* arow = acc + (size_t)d * ld;
    for (int f = threadIdx.x; f < F4; f += blockDim.x) {
        float4 hv = hrow[f];
        atomicAdd(&arow[4 * f + 0], hv.x * w);
        atomicAdd(&arow[4 * f + 1], hv.y * w);
        atomicAdd(&arow[4 * f + 2], hv.z * w);
        atomicAdd(&arow[4 * f + 3], hv.w * w);
    }
}

// in-place: X = relu(acc + H*dinv^2 + bias), zero in padded cols (acc aliases X)
__global__ void conv_epilogue(float* __restrict__ X, const float* __restrict__ H,
                              const float* __restrict__ bias, const float* __restrict__ dinv,
                              int F, int ld, size_t total) {
    size_t idx = (size_t)blockIdx.x * blockDim.x + threadIdx.x;
    if (idx >= total) return;
    int f = (int)(idx % ld);
    size_t i = idx / ld;
    float v = 0.0f;
    if (f < F) {
        float di = dinv[i];
        v = X[idx] + H[idx] * di * di + bias[f];
        v = fmaxf(v, 0.0f);
    }
    X[idx] = v;
}

__global__ void pool_accum(const float* __restrict__ X, const int* __restrict__ batch,
                           float* __restrict__ pool, float* __restrict__ cnt,
                           int F, int ld) {
    int i = blockIdx.x;
    int b = batch[i];
    for (int f = threadIdx.x; f < F; f += blockDim.x)
        atomicAdd(&pool[(size_t)b * ld + f], X[(size_t)i * ld + f]);
    if (threadIdx.x == 0) atomicAdd(&cnt[b], 1.0f);
}

__global__ void pool_div(float* pool, const float* __restrict__ cnt, int ld, size_t total) {
    size_t idx = (size_t)blockIdx.x * blockDim.x + threadIdx.x;
    if (idx >= total) return;
    size_t g = idx / ld;
    pool[idx] /= fmaxf(cnt[g], 1.0f);
}

// W[K x N] fp32 -> Wh[Kpad x Npad] f16 zero padded
__global__ void convert_weight(const float* __restrict__ W, _Float16* __restrict__ Wh,
                               int K, int N, int Npad, int total) {
    int idx = blockIdx.x * blockDim.x + threadIdx.x;
    if (idx >= total) return;
    int k = idx / Npad, n = idx % Npad;
    float v = (k < K && n < N) ? W[(size_t)k * N + n] : 0.0f;
    Wh[idx] = (_Float16)v;
}

__global__ void pad_bias(const float* __restrict__ b, float* __restrict__ bp,
                         int N, int Npad) {
    int n = blockIdx.x * blockDim.x + threadIdx.x;
    if (n < Npad) bp[n] = (n < N) ? b[n] : 0.0f;
}

// out[g] = dot(X[g,:], W) + b  (N-dim = 1 head; not WMMA-friendly)
__global__ void final_out(const float* __restrict__ X, const float* __restrict__ W,
                          const float* __restrict__ b, float* __restrict__ out, int K) {
    int g = blockIdx.x * blockDim.x + threadIdx.x;
    if (g >= N_GRAPHS) return;
    const float* row = X + (size_t)g * K;
    float s = 0.0f;
    for (int k = 0; k < K; ++k) s += row[k] * W[k];
    out[g] = s + b[0];
}

// ---------------------------------------------------------------------------
// Host orchestration
// ---------------------------------------------------------------------------
extern "C" void kernel_launch(void* const* d_in, const int* in_sizes, int n_in,
                              void* d_out, int out_size, void* d_ws, size_t ws_size,
                              hipStream_t stream) {
    (void)in_sizes; (void)n_in; (void)out_size; (void)ws_size;

    // workspace carve-out
    char* p = (char*)d_ws;
    auto take = [&](size_t bytes) -> void* {
        void* r = (void*)p;
        p += (bytes + 255) & ~((size_t)255);
        return r;
    };
    float*    bufX = (float*)take((size_t)N_NODES * 320 * 4);
    float*    bufH = (float*)take((size_t)N_NODES * 320 * 4);
    float*    dinv = (float*)take((size_t)N_NODES * 4);
    float*    pool = (float*)take((size_t)N_GRAPHS * 320 * 4);
    float*    cnt  = (float*)take((size_t)N_GRAPHS * 4);
    float*    gb1  = (float*)take((size_t)N_GRAPHS * 1024 * 4);
    float*    gb2  = (float*)take((size_t)N_GRAPHS * 1024 * 4);
    float*    xc   = (float*)take((size_t)N_GRAPHS * 256 * 4);
    float*    fb1  = (float*)take((size_t)N_GRAPHS * 1024 * 4);
    float*    fb2  = (float*)take((size_t)N_GRAPHS * 512 * 4);
    _Float16* Wh   = (_Float16*)take((size_t)1024 * 1024 * 2);
    float*    bpad = (float*)take(1024 * 4);

    auto run_gemm = [&](const float* A, const float* W, const float* b, float* C,
                        int M, int K, int N, int Kpad, int Npad, int lda, int ldc,
                        bool relu, bool use_bias) {
        int totW = Kpad * Npad;
        convert_weight<<<(totW + 255) / 256, 256, 0, stream>>>(W, Wh, K, N, Npad, totW);
        const float* bp = nullptr;
        if (use_bias) {
            pad_bias<<<(Npad + 255) / 256, 256, 0, stream>>>(b, bpad, N, Npad);
            bp = bpad;
        }
        dim3 grid(M / BM, Npad / BN);
        gemm_wmma_f16<<<grid, 256, 0, stream>>>(A, Wh, C, Kpad, Npad, lda, ldc,
                                                bp, relu ? 1 : 0);
    };

    struct ConvDims { int K, Kp, N, Np, lda; };
    const ConvDims layers[3] = {
        {  78,  96, 156, 192,  96 },
        { 156, 192, 312, 320, 192 },
        { 312, 320, 312, 320, 320 },
    };

    auto run_branch = [&](const float* x, const int* eidx, const int* batch,
                          int pbase, float* xc_out) {
        const int* src = eidx;
        const int* dst = eidx + N_EDGES;

        // deg = 1 (self-loop) + in-degree; then dinv = rsqrt(deg)
        fill_f32<<<(N_NODES + 255) / 256, 256, 0, stream>>>(dinv, 1.0f, N_NODES);
        deg_accum<<<(N_EDGES + 255) / 256, 256, 0, stream>>>(dst, dinv, N_EDGES);
        make_dinv<<<(N_NODES + 255) / 256, 256, 0, stream>>>(dinv, N_NODES);

        // X0: [N_NODES x 96] zero-padded fp32
        {
            size_t tot = (size_t)N_NODES * 96;
            pad_input<<<(unsigned)((tot + 255) / 256), 256, 0, stream>>>(x, bufX, F_IN, 96, tot);
        }

        for (int l = 0; l < 3; ++l) {
            const ConvDims& L = layers[l];
            const float* W = (const float*)d_in[pbase + 2 * l];
            const float* b = (const float*)d_in[pbase + 2 * l + 1];
            // H = X @ W  (WMMA)
            run_gemm(bufX, W, nullptr, bufH, N_NODES, L.K, L.N, L.Kp, L.Np, L.lda, L.Np,
                     false, false);
            // acc (aliases X) = 0; scatter edges; epilogue adds self-loop + bias + relu
            hipMemsetAsync(bufX, 0, (size_t)N_NODES * L.Np * 4, stream);
            scatter_edges<<<N_EDGES, 64, 0, stream>>>(bufH, bufX, src, dst, dinv,
                                                      L.N / 4, L.Np);
            size_t tot = (size_t)N_NODES * L.Np;
            conv_epilogue<<<(unsigned)((tot + 255) / 256), 256, 0, stream>>>(
                bufX, bufH, b, dinv, L.N, L.Np, tot);
        }

        // mean pool to [N_GRAPHS x 320] (cols >= 312 stay zero)
        hipMemsetAsync(pool, 0, (size_t)N_GRAPHS * 320 * 4, stream);
        hipMemsetAsync(cnt, 0, (size_t)N_GRAPHS * 4, stream);
        pool_accum<<<N_NODES, 128, 0, stream>>>(bufX, batch, pool, cnt, 312, 320);
        {
            size_t tot = (size_t)N_GRAPHS * 320;
            pool_div<<<(unsigned)((tot + 255) / 256), 256, 0, stream>>>(pool, cnt, 320, tot);
        }

        // graph MLP: g1 (relu), g2, g4 -> write into xc at column offset
        run_gemm(pool, (const float*)d_in[pbase + 6], (const float*)d_in[pbase + 7],
                 gb1, N_GRAPHS, 312, 1024, 320, 1024, 320, 1024, true, true);
        run_gemm(gb1, (const float*)d_in[pbase + 8], (const float*)d_in[pbase + 9],
                 gb2, N_GRAPHS, 1024, 1024, 1024, 1024, 1024, 1024, false, true);
        run_gemm(gb2, (const float*)d_in[pbase + 10], (const float*)d_in[pbase + 11],
                 xc_out, N_GRAPHS, 1024, 128, 1024, 128, 1024, 256, false, true);
    };

    // solute branch (params 6..17), solvent branch (params 18..29)
    run_branch((const float*)d_in[0], (const int*)d_in[1], (const int*)d_in[2], 6, xc);
    run_branch((const float*)d_in[3], (const int*)d_in[4], (const int*)d_in[5], 18, xc + 128);

    // head: fc1 (relu), fc2 (relu), out (N=1 dot)
    run_gemm(xc, (const float*)d_in[30], (const float*)d_in[31], fb1,
             N_GRAPHS, 256, 1024, 256, 1024, 256, 1024, true, true);
    run_gemm(fb1, (const float*)d_in[32], (const float*)d_in[33], fb2,
             N_GRAPHS, 1024, 512, 1024, 512, 1024, 512, true, true);
    final_out<<<(N_GRAPHS + 255) / 256, 256, 0, stream>>>(
        fb2, (const float*)d_in[34], (const float*)d_in[35], (float*)d_out, 512);
}
